// PrototypeModel_34213709480382
// MI455X (gfx1250) — compile-verified, matrix operand order checked by reading, not run on hardware
//
#include <hip/hip_runtime.h>

// yat-similarity: scores[i,j] = <x_i,p_j>^2 / (||x_i - p_j||^2 + eps)
// GEMM 16384x1024x1024 is compute-bound (AI ~260 FLOP/B vs 23.3 TB/s HBM).
// Precision lane: split-precision bf16 (x = hi + lo, dot = hi*hi + hi*lo + lo*hi,
// f32 accumulate) -> 3x V_WMMA_F32_16X16X32_BF16 per 16x16xK32 tile instead of
// 8x V_WMMA_F32_16X16X4_F32, each on the fast 16-bit XDL pipe. Rel. error of the
// dot ~3e-5 (fp32-class); norms and epilogue stay exact f32.

typedef float  v2f  __attribute__((ext_vector_type(2)));
typedef float  v4f  __attribute__((ext_vector_type(4)));
typedef float  v8f  __attribute__((ext_vector_type(8)));
typedef __bf16 v4bf __attribute__((ext_vector_type(4)));
typedef __bf16 v8bf __attribute__((ext_vector_type(8)));
typedef __bf16 v16bf __attribute__((ext_vector_type(16)));

#define BM   128
#define BN   128
#define KT   32
#define LDSB 40    // bf16 elems per LDS row (80B): 16B-aligned rows, 20*M mod 64 -> 16 banks
#define DDIM 1024

// A/B 16-bit fragment (16x32): lanes 0-15 M=row, VGPR0-3: K=0..7, VGPR4-7: K=16..23;
// lanes 16-31: K=8..15 and K=24..31.  rowp points at LDS row start.
__device__ __forceinline__ v16bf load_frag(const __bf16* rowp, int hi16) {
    v8bf a = *(const v8bf*)(rowp + 8 * hi16);
    v8bf b = *(const v8bf*)(rowp + 16 + 8 * hi16);
    return __builtin_shufflevector(a, b, 0, 1, 2, 3, 4, 5, 6, 7,
                                         8, 9, 10, 11, 12, 13, 14, 15);
}

__global__ __launch_bounds__(256)
void yat_scores_kernel(const float* __restrict__ X,
                       const float* __restrict__ Pm,
                       float* __restrict__ Out)
{
    __shared__ __bf16 Xhi[BM * LDSB];
    __shared__ __bf16 Xlo[BM * LDSB];
    __shared__ __bf16 Phi[BN * LDSB];
    __shared__ __bf16 Plo[BN * LDSB];
    __shared__ float  xsq_sh[BM];
    __shared__ float  psq_sh[BN];

    const int t    = threadIdx.x;
    const int lane = t & 31;
    const int hi16 = lane >> 4;          // 0 / 1
    const int l16  = lane & 15;
    const int wave = t >> 5;             // 0..7
    const int waveM = (wave >> 2) * 64;  // 0 or 64
    const int waveN = (wave & 3) * 32;   // 0,32,64,96

    const int mpanel = blockIdx.y;
    const int npanel = blockIdx.x;
    const long long xbase = (long long)mpanel * BM * DDIM;
    const long long pbase = (long long)npanel * BN * DDIM;

    // loader mapping: 8 threads/row (4 consecutive K each), 32 rows/pass, 4 passes
    const int lrow = t >> 3;        // 0..31
    const int lcol = (t & 7) * 4;   // 0,4,...,28

    v8f acc[4][2];
    #pragma unroll
    for (int mi = 0; mi < 4; ++mi)
        #pragma unroll
        for (int ni = 0; ni < 2; ++ni)
            acc[mi][ni] = (v8f){};

    float xacc[4] = {0.f, 0.f, 0.f, 0.f};
    float pacc[4] = {0.f, 0.f, 0.f, 0.f};

    v4f xv[4], pv[4];
    // prefetch tile 0
    #pragma unroll
    for (int p = 0; p < 4; ++p) {
        const int r = p * 32 + lrow;
        xv[p] = *(const v4f*)(X  + xbase + (long long)r * DDIM + lcol);
        pv[p] = *(const v4f*)(Pm + pbase + (long long)r * DDIM + lcol);
    }

    for (int kt = 0; kt < DDIM; kt += KT) {
        __syncthreads();   // previous iteration's fragment reads done
        // stage: f32 -> bf16 hi/lo into LDS, fused squared-norm partials
        #pragma unroll
        for (int p = 0; p < 4; ++p) {
            const int r = p * 32 + lrow;
            const v4f v = xv[p];
            const v4f w = pv[p];
            __bf16 h0 = (__bf16)v.x, h1 = (__bf16)v.y, h2 = (__bf16)v.z, h3 = (__bf16)v.w;
            __bf16 g0 = (__bf16)w.x, g1 = (__bf16)w.y, g2 = (__bf16)w.z, g3 = (__bf16)w.w;
            *(v4bf*)&Xhi[r * LDSB + lcol] = (v4bf){h0, h1, h2, h3};
            *(v4bf*)&Phi[r * LDSB + lcol] = (v4bf){g0, g1, g2, g3};
            *(v4bf*)&Xlo[r * LDSB + lcol] = (v4bf){(__bf16)(v.x - (float)h0),
                                                   (__bf16)(v.y - (float)h1),
                                                   (__bf16)(v.z - (float)h2),
                                                   (__bf16)(v.w - (float)h3)};
            *(v4bf*)&Plo[r * LDSB + lcol] = (v4bf){(__bf16)(w.x - (float)g0),
                                                   (__bf16)(w.y - (float)g1),
                                                   (__bf16)(w.z - (float)g2),
                                                   (__bf16)(w.w - (float)g3)};
            xacc[p] += v.x * v.x + v.y * v.y + v.z * v.z + v.w * v.w;
            pacc[p] += w.x * w.x + w.y * w.y + w.z * w.z + w.w * w.w;
        }
        __syncthreads();

        // software pipeline: issue next tile's global loads before compute
        if (kt + KT < DDIM) {
            #pragma unroll
            for (int p = 0; p < 4; ++p) {
                const int r = p * 32 + lrow;
                xv[p] = *(const v4f*)(X  + xbase + (long long)r * DDIM + (kt + KT) + lcol);
                pv[p] = *(const v4f*)(Pm + pbase + (long long)r * DDIM + (kt + KT) + lcol);
            }
        }

        // B fragments for the whole K=32 chunk
        v16bf bhi[2], blo[2];
        #pragma unroll
        for (int ni = 0; ni < 2; ++ni) {
            const int row = waveN + ni * 16 + l16;
            bhi[ni] = load_frag(&Phi[row * LDSB], hi16);
            blo[ni] = load_frag(&Plo[row * LDSB], hi16);
        }
        #pragma unroll
        for (int mi = 0; mi < 4; ++mi) {
            const int row = waveM + mi * 16 + l16;
            const v16bf ahi = load_frag(&Xhi[row * LDSB], hi16);
            const v16bf alo = load_frag(&Xlo[row * LDSB], hi16);
            #pragma unroll
            for (int ni = 0; ni < 2; ++ni) {
                acc[mi][ni] = __builtin_amdgcn_wmma_f32_16x16x32_bf16(
                    false, ahi, false, bhi[ni], (short)0, acc[mi][ni], false, false);
                acc[mi][ni] = __builtin_amdgcn_wmma_f32_16x16x32_bf16(
                    false, ahi, false, blo[ni], (short)0, acc[mi][ni], false, false);
                acc[mi][ni] = __builtin_amdgcn_wmma_f32_16x16x32_bf16(
                    false, alo, false, bhi[ni], (short)0, acc[mi][ni], false, false);
            }
        }
    }

    // reduce squared-norm partials across the 8 loader lanes per row
    #pragma unroll
    for (int p = 0; p < 4; ++p) {
        float xs = xacc[p], ps = pacc[p];
        #pragma unroll
        for (int off = 4; off >= 1; off >>= 1) {
            xs += __shfl_down(xs, off, 8);
            ps += __shfl_down(ps, off, 8);
        }
        if ((t & 7) == 0) {
            xsq_sh[p * 32 + lrow] = xs;
            psq_sh[p * 32 + lrow] = ps;
        }
    }
    __syncthreads();

    // fused rational epilogue; C layout: VGPR v -> M = v + 8*hi16, lane -> N = l16
    const long long obase = (long long)(mpanel * BM) * DDIM + npanel * BN;
    #pragma unroll
    for (int mi = 0; mi < 4; ++mi) {
        #pragma unroll
        for (int ni = 0; ni < 2; ++ni) {
            const v8f c = acc[mi][ni];
            const int nloc = waveN + ni * 16 + l16;
            const float ps = psq_sh[nloc];
            #pragma unroll
            for (int v = 0; v < 8; ++v) {
                const int mloc = waveM + mi * 16 + v + 8 * hi16;
                const float dot = c[v];
                const float xs  = xsq_sh[mloc];
                const float den = xs + ps - 2.0f * dot + 1e-6f;
                Out[obase + (long long)mloc * DDIM + nloc] = dot * dot / den;
            }
        }
    }
}

extern "C" void kernel_launch(void* const* d_in, const int* in_sizes, int n_in,
                              void* d_out, int out_size, void* d_ws, size_t ws_size,
                              hipStream_t stream) {
    (void)n_in; (void)out_size; (void)d_ws; (void)ws_size;
    const float* X  = (const float*)d_in[0];
    const float* Pm = (const float*)d_in[1];
    float* Out = (float*)d_out;

    const int D = DDIM;
    const int Brows = in_sizes[0] / D;   // 16384
    const int Pn    = in_sizes[1] / D;   // 1024

    dim3 grid(Pn / BN, Brows / BM);      // (8, 128)
    hipLaunchKernelGGL(yat_scores_kernel, grid, dim3(256), 0, stream, X, Pm, Out);
}
